// DSGCNBlock_2723009265867
// MI455X (gfx1250) — matrix operand order repaired and checked by reading, not compiled, not run on hardware
//
#include <hip/hip_runtime.h>
#include <hip/hip_bf16.h>
#include <math.h>

typedef __attribute__((ext_vector_type(16))) _Float16 v16h;
typedef __attribute__((ext_vector_type(8)))  float    v8f;

// Problem dims (fixed by the reference)
#define BB   16
#define TT   128
#define NN   64
#define CIN  128
#define COUT 256
#define KK   3
#define EPS  1e-5f

// ---------------------------------------------------------------------------
// P1: s[k][n] = sum_m A[k][n][m]   (192 outputs)
// ---------------------------------------------------------------------------
__global__ void k_rowsum(const float* __restrict__ A, float* __restrict__ s) {
    int i = blockIdx.x * blockDim.x + threadIdx.x;
    if (i < KK * NN) {
        const float* row = A + (size_t)i * NN;
        float acc = 0.f;
        #pragma unroll
        for (int m = 0; m < NN; ++m) acc += row[m];
        s[i] = acc;
    }
}

// ---------------------------------------------------------------------------
// P2: build f16 weight fragments in WMMA B-operand order.
// Layout: wf[n][ks(4)][otile(32)][lane(32)][e(16)]  (16 halfs = 32B per lane)
// B 32x16 mapping: column = lane%16 ; K = ks*32 + (lane<16 ? 0 : 16) + e
// otile 0..15  -> W_eff (per-node folded pointwise weight, cols 0..255)
// otile 16..31 -> W_res (residual projection, cols 256..511)
// ---------------------------------------------------------------------------
__global__ void k_build_wfrag(const float* __restrict__ s,
                              const float* __restrict__ dw,
                              const float* __restrict__ Wpw,
                              const float* __restrict__ Wres,
                              _Float16* __restrict__ wf) {
    int idx = blockIdx.x * 256 + threadIdx.x;          // 0 .. 4194303
    int e     =  idx        & 15;
    int lane  = (idx >> 4)  & 31;
    int otile = (idx >> 9)  & 31;
    int ks    = (idx >> 14) & 3;
    int n     =  idx >> 16;                            // 0..63
    int o = otile * 16 + (lane & 15);                  // 0..511
    int k = ks * 32 + ((lane < 16) ? 0 : 16) + e;      // 0..127
    float v;
    if (o < COUT) {
        v = 0.f;
        #pragma unroll
        for (int kk = 0; kk < KK; ++kk)
            v += s[kk * NN + n] * dw[kk * CIN + k] * Wpw[(size_t)o * (KK * CIN) + kk * CIN + k];
    } else {
        v = Wres[(size_t)(o - COUT) * CIN + k];
    }
    wf[idx] = (_Float16)v;
}

// ---------------------------------------------------------------------------
// G: fused GEMM. For each node n, rows = (b,t), K = 128, cols = 512
//    cols [0,256)  -> h2   (workspace)
//    cols [256,512)-> residual (stashed in d_out)
// Block = 256 threads (8 waves). Block handles one (n, 16-row tile).
// Each wave: 4 column tiles of 16, K-chain of 4x v_wmma_f32_16x16x32_f16.
// ---------------------------------------------------------------------------
__global__ void k_gemm(const float* __restrict__ x,
                       const _Float16* __restrict__ wf,
                       float* __restrict__ h2,
                       float* __restrict__ res) {
    __shared__ __align__(32) _Float16 ldsA[4 * 32 * 16];   // [ks][lane][e]

    int blk = blockIdx.x;
    int n  = blk >> 7;                 // 0..63
    int rt = blk & 127;                // row tile within node
    int b  = rt >> 3;                  // 0..15
    int t0 = (rt & 7) << 4;            // 0,16,...,112
    int tid = threadIdx.x;

    // ---- stage A tile (16 rows x 128 K, fp32 -> f16) into fragment order ----
    {
        int m  = tid >> 4;             // row 0..15
        int kb = (tid & 15) << 3;      // 8 consecutive k per thread
        const float* src = x + (((size_t)(b * TT + t0 + m) * NN + n) << 7) + kb;
        #pragma unroll
        for (int j = 0; j < 8; ++j) {
            int k  = kb + j;
            int ks = k >> 5, kk = k & 31;
            int lane, e;
            if      (kk < 8)  { lane = m;      e = kk;      }
            else if (kk < 16) { lane = m;      e = kk;      }
            else if (kk < 24) { lane = m + 16; e = kk - 16; }
            else              { lane = m + 16; e = kk - 16; }
            ldsA[(ks << 9) + (lane << 4) + e] = (_Float16)src[j];
        }
    }
    __syncthreads();

    int w = tid >> 5, lane = tid & 31;
    v16h a[4];
    #pragma unroll
    for (int ks = 0; ks < 4; ++ks)
        a[ks] = *(const v16h*)&ldsA[(ks << 9) + (lane << 4)];

    const _Float16* wfn = wf + ((size_t)n << 16);   // 65536 halfs per node

    #pragma unroll
    for (int ct = 0; ct < 4; ++ct) {
        int otile = (w << 2) + ct;                  // 0..31 (uniform per wave)
        v8f c = {};
        #pragma unroll
        for (int ks = 0; ks < 4; ++ks) {
            v16h bf = *(const v16h*)&wfn[(size_t)(((ks * 32 + otile) * 32 + lane)) << 4];
            c = __builtin_amdgcn_wmma_f32_16x16x32_f16(
                    false, a[ks], false, bf, (short)0, c, false, false);
        }
        // D layout: VGPR r -> M = r (lanes 0-15) / r+8 (lanes 16-31), N = lane%16
        int colbase = otile << 4;
        int col = (colbase & 255) + (lane & 15);
        float* dst = (otile < 16) ? h2 : res;
        int trow0 = t0 + ((lane < 16) ? 0 : 8);
        #pragma unroll
        for (int r = 0; r < 8; ++r) {
            int t = trow0 + r;
            dst[(((size_t)(b * TT + t) * NN + n) << 8) + col] = c[r];
        }
    }
}

// ---------------------------------------------------------------------------
// C1: depthwise temporal conv (3 taps, zero pad) + GroupNorm statistics.
// One block per (b, n, g): mean/rstd over 32 channels x 128 timesteps.
// ---------------------------------------------------------------------------
__global__ void k_gnstats(const float* __restrict__ h2,
                          const float* __restrict__ wconv,
                          float* __restrict__ stats) {
    __shared__ float red[512];
    int blk = blockIdx.x;              // ((b*64)+n)*8 + g
    int g = blk & 7;
    int n = (blk >> 3) & 63;
    int b = blk >> 9;
    int tid = threadIdx.x;
    int o = (g << 5) + (tid & 31);
    int tbase = (tid >> 5) << 4;       // 16 timesteps per thread

    float w0 = wconv[o * 3], w1 = wconv[o * 3 + 1], w2 = wconv[o * 3 + 2];
    const size_t stT = (size_t)NN * COUT;
    size_t base = ((size_t)b * TT * NN + n) * COUT + o;

    float sum = 0.f, sq = 0.f;
    #pragma unroll
    for (int j = 0; j < 16; ++j) {
        int t = tbase + j;
        float xm = (t > 0)      ? h2[base + (size_t)(t - 1) * stT] : 0.f;
        float xc =                h2[base + (size_t)t * stT];
        float xp = (t < TT - 1) ? h2[base + (size_t)(t + 1) * stT] : 0.f;
        float v = w0 * xm + w1 * xc + w2 * xp;
        sum += v; sq += v * v;
    }
    red[tid] = sum; red[256 + tid] = sq;
    __syncthreads();
    for (int st = 128; st > 0; st >>= 1) {
        if (tid < st) { red[tid] += red[tid + st]; red[256 + tid] += red[256 + tid + st]; }
        __syncthreads();
    }
    if (tid == 0) {
        float mean = red[0] * (1.f / 4096.f);
        float var  = red[256] * (1.f / 4096.f) - mean * mean;
        stats[blk * 2]     = mean;
        stats[blk * 2 + 1] = rsqrtf(var + EPS);
    }
}

// ---------------------------------------------------------------------------
// F: conv (recomputed) -> GroupNorm -> +residual -> LayerNorm(256) -> GELU.
// One block per (b,t,n) row, 256 threads = one output channel each.
// Residual is read from d_out (stashed by k_gemm), then overwritten.
// ---------------------------------------------------------------------------
__global__ void k_final(const float* __restrict__ h2,
                        const float* __restrict__ stats,
                        const float* __restrict__ wconv,
                        const float* __restrict__ gnw, const float* __restrict__ gnb,
                        const float* __restrict__ lnw, const float* __restrict__ lnb,
                        float* out) {
    __shared__ float red[512];
    int row = blockIdx.x;              // (b*128+t)*64 + n
    int n = row & 63;
    int t = (row >> 6) & 127;
    int b = row >> 13;
    int o = threadIdx.x;

    const size_t stT = (size_t)NN * COUT;   // t+-1 moves row by +-64 -> +-stT floats
    size_t base = (size_t)row * COUT + o;

    float w0 = wconv[o * 3], w1 = wconv[o * 3 + 1], w2 = wconv[o * 3 + 2];
    float xm = (t > 0)      ? h2[base - stT] : 0.f;
    float xc =                h2[base];
    float xp = (t < TT - 1) ? h2[base + stT] : 0.f;
    float conv = w0 * xm + w1 * xc + w2 * xp;

    int g = o >> 5;
    int sidx = (((b * NN + n) << 3) + g) * 2;
    float gn = (conv - stats[sidx]) * stats[sidx + 1] * gnw[o] + gnb[o];
    float v = gn + out[base];          // residual

    red[o] = v; red[256 + o] = v * v;
    __syncthreads();
    for (int st = 128; st > 0; st >>= 1) {
        if (o < st) { red[o] += red[o + st]; red[256 + o] += red[256 + o + st]; }
        __syncthreads();
    }
    float mu  = red[0]   * (1.f / 256.f);
    float var = red[256] * (1.f / 256.f) - mu * mu;
    float y = (v - mu) * rsqrtf(var + EPS) * lnw[o] + lnb[o];
    out[base] = 0.5f * y * (1.f + erff(y * 0.70710678118654752f));
}

// ---------------------------------------------------------------------------
extern "C" void kernel_launch(void* const* d_in, const int* in_sizes, int n_in,
                              void* d_out, int out_size, void* d_ws, size_t ws_size,
                              hipStream_t stream) {
    const float* x     = (const float*)d_in[0];
    const float* A     = (const float*)d_in[1];
    const float* dw    = (const float*)d_in[2];
    const float* Wpw   = (const float*)d_in[3];
    const float* Wconv = (const float*)d_in[4];
    const float* gnw   = (const float*)d_in[5];
    const float* gnb   = (const float*)d_in[6];
    const float* lnw   = (const float*)d_in[7];
    const float* lnb   = (const float*)d_in[8];
    const float* Wres  = (const float*)d_in[9];
    float* out = (float*)d_out;

    char* ws = (char*)d_ws;
    float*    s     = (float*)(ws + 0);                       // 768 B
    float*    stats = (float*)(ws + 1024);                    // 64 KB
    _Float16* wfrag = (_Float16*)(ws + 131072);               // 8 MB
    float*    h2    = (float*)(ws + (size_t)16 * 1024 * 1024);// 128 MB

    k_rowsum     <<<1, 192, 0, stream>>>(A, s);
    k_build_wfrag<<<(64 * 4 * 32 * 32 * 16) / 256, 256, 0, stream>>>(s, dw, Wpw, Wres, wfrag);
    k_gemm       <<<NN * (BB * TT / 16), 256, 0, stream>>>(x, wfrag, h2, out);
    k_gnstats    <<<BB * NN * 8, 256, 0, stream>>>(h2, Wconv, stats);
    k_final      <<<BB * TT * NN, 256, 0, stream>>>(h2, stats, Wconv, gnw, gnb, lnw, lnb, out);
}
